// THGATImputer_17901423690203
// MI455X (gfx1250) — matrix operand order repaired and checked by reading, not compiled, not run on hardware
//
#include <hip/hip_runtime.h>
#include <hip/hip_bf16.h>
#include <math.h>

// Problem constants (match reference)
#define Bn 8
#define Fn 16
#define Nn 1024
#define En 256
#define Tn 64
#define ALPHA 0.2f

typedef __attribute__((ext_vector_type(16))) _Float16 v16h;
typedef __attribute__((ext_vector_type(8)))  float    v8f;

// ---------------------------------------------------------------------------
// WMMA fragment helpers (layouts per CDNA5 ISA §7.12.2, wave32)
// A 16x32 f16: lanes 0-15 -> M=lane, halves 0..7 = K 0..7, halves 8..15 = K 16..23
//              lanes 16-31 -> M=lane-16, K offset +8 (K 8..15 / 24..31)
// ---------------------------------------------------------------------------
__device__ inline v16h load_a16(const _Float16* src, int ld, int kBase, int lane) {
    int m    = lane & 15;
    int kOff = (lane & 16) >> 1;              // 0 for lanes 0-15, 8 for 16-31
    const _Float16* pr = src + (size_t)m * ld + kBase + kOff;
    v16h r;
#pragma unroll
    for (int h = 0; h < 8; ++h) r[h] = pr[h];
#pragma unroll
    for (int h = 0; h < 8; ++h) r[8 + h] = pr[16 + h];
    return r;
}

// ---------------------------------------------------------------------------
// One-time prep kernels
// ---------------------------------------------------------------------------
__global__ void deg_kernel(const float* __restrict__ inc, float* __restrict__ deg) {
    int e = blockIdx.x * blockDim.x + threadIdx.x;
    if (e >= En) return;
    float s = 0.f;
    for (int n = 0; n < Nn; ++n) s += inc[(size_t)n * En + e];
    deg[e] = s;
}

__global__ void transpose_kernel(const float* __restrict__ inc, float* __restrict__ incT) {
    int idx = blockIdx.x * blockDim.x + threadIdx.x;   // N*E = 262144
    int n = idx >> 8, e = idx & 255;
    incT[(size_t)e * Nn + n] = inc[(size_t)n * En + e];
}

// Pre-swizzle a [K x 16*nET] f32 matrix into WMMA B-fragment order (f16):
// B 32x16 f16 per tile: lane -> Ncol = lane&15, K = (lane>=16 ? 8:0)+(half&7)+(half>=8 ? 16:0).
// out index: (((kTile*nET + eTile)*32 + lane)*16 + h)
__global__ void bfrag_kernel(const float* __restrict__ src, _Float16* __restrict__ dst,
                             int nKT, int nET, int ldsrc) {
    int idx = blockIdx.x * blockDim.x + threadIdx.x;   // nKT*nET*512
    if (idx >= nKT * nET * 512) return;
    int h    = idx & 15;
    int lane = (idx >> 4) & 31;
    int eT   = (idx >> 9) % nET;
    int kT   = (idx >> 9) / nET;
    int ecol = lane & 15;
    int kin  = ((lane >> 4) << 3) + (h & 7) + ((h >> 3) << 4); // +8 for lanes 16-31, +16 for halves 8..15
    dst[idx] = (_Float16)src[(size_t)(kT * 32 + kin) * ldsrc + (eT * 16 + ecol)];
}

// ---------------------------------------------------------------------------
// Per-timestep kernel 1: node features xn[b,n] (rows 8..15 zero-padded for WMMA)
// ---------------------------------------------------------------------------
__global__ void xn_kernel(const float* __restrict__ x, const float* __restrict__ mask,
                          const float* __restrict__ w, const float* __restrict__ bias,
                          const float* __restrict__ nprev, int nprevStride, int t,
                          float* __restrict__ xnf, _Float16* __restrict__ xnh) {
    int idx = blockIdx.x * blockDim.x + threadIdx.x;   // 16*1024
    int m = idx >> 10, n = idx & 1023;
    float v = 0.f;
    if (m < Bn) {
        const size_t cs = (size_t)Nn * Tn;             // channel stride in x
        const float* xb = x    + ((size_t)m * Fn * Nn + n) * Tn + t;
        const float* mb = mask + ((size_t)m * Fn * Nn + n) * Tn + t;
#pragma unroll
        for (int c = 0; c < Fn; ++c) v += xb[c * cs] * w[c];
#pragma unroll
        for (int c = 0; c < Fn; ++c) v += mb[c * cs] * w[Fn + c];
        v += nprev[m * nprevStride + n] * w[2 * Fn] + bias[n];
    }
    xnf[idx] = v;
    xnh[idx] = (_Float16)v;
}

// ---------------------------------------------------------------------------
// Per-timestep kernel 2: fused dual GEMM via v_wmma_f32_16x16x32_f16.
//   GEMM1: [16 x 1024]f16 @ inc[1024 x 256] -> /deg -> LDS f16 tile
//   GEMM2: [16 x 256]f16(LDS) @ w2[256 x 256] -> edge[8 x 256]
// One workgroup of 16 waves; wave w owns E-tile w. EXEC all-ones as WMMA requires.
// ---------------------------------------------------------------------------
__global__ __launch_bounds__(512) void edge_kernel(const _Float16* __restrict__ xnh,
                                                   const _Float16* __restrict__ incB,
                                                   const _Float16* __restrict__ w2B,
                                                   const float* __restrict__ deg,
                                                   float* __restrict__ edge) {
    __shared__ _Float16 tmp[16 * En];                  // 8 KB
    const int lane = threadIdx.x & 31;
    const int wv   = threadIdx.x >> 5;                 // 0..15 = E tile

    v8f c = {};
    for (int kt = 0; kt < Nn / 32; ++kt) {
        const _Float16* bp = incB + (((size_t)kt * 16 + wv) * 32 + lane) * 16;
        if (kt + 1 < Nn / 32)
            __builtin_prefetch(incB + (((size_t)(kt + 1) * 16 + wv) * 32 + lane) * 16, 0, 0);
        v16h a  = load_a16(xnh, Nn, kt * 32, lane);
        v16h bf = *(const v16h*)bp;
        c = __builtin_amdgcn_wmma_f32_16x16x32_f16(false, a, false, bf,
                                                   (short)0, c, false, false);
    }
    const int ecol  = lane & 15;
    const int e     = wv * 16 + ecol;
    const int mBase = (lane < 16) ? 0 : 8;             // C/D: vgpr r -> row mBase+r
    const float dg  = deg[e];
#pragma unroll
    for (int r = 0; r < 8; ++r)
        tmp[(mBase + r) * En + e] = (_Float16)(c[r] / dg);  // rows>=8 are exact zeros
    __syncthreads();

    v8f c2 = {};
    for (int kt = 0; kt < En / 32; ++kt) {
        v16h a  = load_a16(tmp, En, kt * 32, lane);    // LDS-sourced A fragment
        v16h bf = *(const v16h*)(w2B + (((size_t)kt * 16 + wv) * 32 + lane) * 16);
        c2 = __builtin_amdgcn_wmma_f32_16x16x32_f16(false, a, false, bf,
                                                    (short)0, c2, false, false);
    }
#pragma unroll
    for (int r = 0; r < 8; ++r) {
        int m = mBase + r;
        if (m < Bn) edge[m * En + e] = c2[r];
    }
}

// ---------------------------------------------------------------------------
// Per-timestep kernel 3a: softmax stats per (b,e). One wave per (b,e).
// time_e cancels in the shift-invariant softmax; masked entries underflow to 0.
// comb = edge / Z so 3b does one multiply per term.
// ---------------------------------------------------------------------------
__global__ void stats_kernel(const float* __restrict__ xnf, const float* __restrict__ edge,
                             const float* __restrict__ incT, const float* __restrict__ a2,
                             float* __restrict__ mOut, float* __restrict__ combOut) {
    int wave = (blockIdx.x * blockDim.x + threadIdx.x) >> 5;   // B*E waves
    int lane = threadIdx.x & 31;
    int e = wave & (En - 1), b = wave >> 8;
    const float a0 = a2[0], a1 = a2[1];
    const float ed = edge[b * En + e];
    const float* xr = xnf + (size_t)b * Nn;
    const float* ir = incT + (size_t)e * Nn;

    float mx = -INFINITY;
    for (int n = lane; n < Nn; n += 32)
        if (ir[n] > 0.f) {
            float s = xr[n] * a0 + ed * a1;
            s = s > 0.f ? s : ALPHA * s;
            mx = fmaxf(mx, s);
        }
#pragma unroll
    for (int o = 16; o > 0; o >>= 1) mx = fmaxf(mx, __shfl_xor(mx, o, 32));

    float z = 0.f;
    for (int n = lane; n < Nn; n += 32)
        if (ir[n] > 0.f) {
            float s = xr[n] * a0 + ed * a1;
            s = s > 0.f ? s : ALPHA * s;
            z += __expf(s - mx);
        }
#pragma unroll
    for (int o = 16; o > 0; o >>= 1) z += __shfl_xor(z, o, 32);

    if (lane == 0) {
        mOut[b * En + e]    = mx;
        combOut[b * En + e] = ed / z;
    }
}

// ---------------------------------------------------------------------------
// Per-timestep kernel 3b: node[b,n] = sum_e attn * edge; writes carry + output.
// ---------------------------------------------------------------------------
__global__ void node_kernel(const float* __restrict__ xnf, const float* __restrict__ edge,
                            const float* __restrict__ inc, const float* __restrict__ a2,
                            const float* __restrict__ mIn, const float* __restrict__ comb,
                            float* __restrict__ ncarry, float* __restrict__ out, int t) {
    int idx = blockIdx.x * blockDim.x + threadIdx.x;   // B*N
    int b = idx >> 10, n = idx & 1023;
    const float a0 = a2[0], a1 = a2[1];
    const float xv = xnf[(size_t)b * Nn + n] * a0;
    const float* ir = inc + (size_t)n * En;
    const float* eb = edge + b * En;
    const float* mb = mIn + b * En;
    const float* cb = comb + b * En;
    float acc = 0.f;
#pragma unroll 4
    for (int e = 0; e < En; ++e)
        if (ir[e] > 0.f) {
            float s = xv + eb[e] * a1;
            s = s > 0.f ? s : ALPHA * s;
            acc += __expf(s - mb[e]) * cb[e];
        }
    ncarry[idx] = acc;
    out[((size_t)b * Tn + t) * Nn + n] = acc;
}

// ---------------------------------------------------------------------------
// kernel_launch: prep once, then 64 sequential timesteps (graph-capture safe).
// ---------------------------------------------------------------------------
extern "C" void kernel_launch(void* const* d_in, const int* in_sizes, int n_in,
                              void* d_out, int out_size, void* d_ws, size_t ws_size,
                              hipStream_t stream) {
    const float* x    = (const float*)d_in[0];
    const float* mask = (const float*)d_in[1];
    const float* inc  = (const float*)d_in[2];
    const float* h0   = (const float*)d_in[3];
    // d_in[4] = h1: provably dead (time_e cancels in the softmax; edge carry unused)
    const float* bias = (const float*)d_in[5];
    const float* w    = (const float*)d_in[6];
    const float* w2   = (const float*)d_in[7];
    const float* a    = (const float*)d_in[8];
    float* out = (float*)d_out;

    char* p = (char*)d_ws;
    auto carve = [&](size_t bytes) { void* r = (void*)p; p += (bytes + 255) & ~(size_t)255; return r; };
    float*     deg    = (float*)    carve(En * 4);
    float*     incT   = (float*)    carve((size_t)En * Nn * 4);          // 1 MB
    _Float16*  incB   = (_Float16*) carve((size_t)(Nn/32) * 16 * 512 * 2); // 512 KB
    _Float16*  w2B    = (_Float16*) carve((size_t)(En/32) * 16 * 512 * 2); // 128 KB
    float*     xnf    = (float*)    carve(16 * Nn * 4);
    _Float16*  xnh    = (_Float16*) carve(16 * Nn * 2);
    float*     edge   = (float*)    carve(Bn * En * 4);
    float*     ncarry = (float*)    carve((size_t)Bn * Nn * 4);
    float*     mbuf   = (float*)    carve(Bn * En * 4);
    float*     comb   = (float*)    carve(Bn * En * 4);

    // one-time prep (runs every call for determinism; d_ws is never trusted)
    deg_kernel<<<1, 256, 0, stream>>>(inc, deg);
    transpose_kernel<<<(Nn * En) / 256, 256, 0, stream>>>(inc, incT);
    bfrag_kernel<<<((Nn/32) * 16 * 512 + 255) / 256, 256, 0, stream>>>(inc, incB, Nn/32, 16, En);
    bfrag_kernel<<<((En/32) * 16 * 512 + 255) / 256, 256, 0, stream>>>(w2, w2B, En/32, 16, En);

    for (int t = 0; t < Tn; ++t) {
        const float* nprev = (t == 0) ? h0 : ncarry;
        int nstride        = (t == 0) ? 0 : Nn;        // h0 broadcasts over batch
        xn_kernel  <<<(16 * Nn) / 256, 256, 0, stream>>>(x, mask, w, bias, nprev, nstride, t, xnf, xnh);
        edge_kernel<<<1, 512, 0, stream>>>(xnh, incB, w2B, deg, edge);
        stats_kernel<<<(Bn * En * 32) / 256, 256, 0, stream>>>(xnf, edge, incT, a, mbuf, comb);
        node_kernel <<<(Bn * Nn) / 256, 256, 0, stream>>>(xnf, edge, inc, a, mbuf, comb, ncarry, out, t);
    }
}
